// NMTDecoder_30958124269571
// MI455X (gfx1250) — matrix operand order repaired and checked by reading, not compiled, not run on hardware
//
#include <hip/hip_runtime.h>
#include <math.h>

#define BB  32      // batch
#define SS  64      // source length
#define TT  64      // target length
#define HH  1024    // hidden
#define EE  512     // embed
#define VV  32000   // vocab
#define G3H 3072    // 3*H
#define KIH 1536    // E+H
#define K2H 2048    // 2*H

typedef __attribute__((ext_vector_type(16))) __bf16 v16bf;
typedef __attribute__((ext_vector_type(8)))  float  v8f;

#define WMMA_BF16(a, b, c) \
  __builtin_amdgcn_wmma_f32_16x16x32_bf16(false, (a), false, (b), (short)0, (c), false, false)

// ---------- helpers ----------
__device__ __forceinline__ unsigned short f2bf(float f) {
  unsigned int u = __builtin_bit_cast(unsigned int, f);
  u += 0x7FFFu + ((u >> 16) & 1u);            // round-to-nearest-even
  return (unsigned short)(u >> 16);
}
__device__ __forceinline__ float sigm(float x) { return 1.0f / (1.0f + expf(-x)); }

// A operand (16x32 MxK bf16), per ISA 7.12.2: lane L -> row m = L&15;
// VGPR v, half h -> k = (v>=4?16:0) + (v&3)*2 + 8*(L>>4) + h.
// Fragment element index within tile: ((kb*32 + lane)*16 + e), e = v*2+h.
__device__ __forceinline__ void packA_decomp(int tid, int KB, int& m, int& k, int& mt) {
  int e    = tid & 15;
  int lane = (tid >> 4) & 31;
  int kb   = (tid >> 9) % KB;
  mt       = tid / (KB << 9);
  int v = e >> 1, h = e & 1;
  int kl = ((v >= 4) ? 16 : 0) + ((v & 3) << 1) + ((lane >> 4) << 3) + h;
  m = mt * 16 + (lane & 15);
  k = kb * 32 + kl;
}

// ---------- weight swizzle into B-operand fragments ----------
// B operand (32x16 KxN bf16): lane L -> col n = L&15, K-half = L>>4.
// Per lane e -> k = (e>=8?16:0) + 8*(L>>4) + (e&7) (matches A); storage ((nt*KB + kb)*32 + lane)*16 + e.
__global__ void k_swizzleB(const float* __restrict__ W, unsigned short* __restrict__ out,
                           int KB, int total) {
  int tid = blockIdx.x * blockDim.x + threadIdx.x;
  if (tid >= total) return;
  int e    = tid & 15;
  int lane = (tid >> 4) & 31;
  int kb   = (tid >> 9) % KB;
  int nt   = tid / (KB << 9);
  int K    = KB << 5;
  int n    = nt * 16 + (lane & 15);
  int k    = kb * 32 + ((e >> 3) << 4) + ((lane >> 4) << 3) + (e & 7);
  out[tid] = f2bf(W[(size_t)n * K + k]);
}

// ---------- fused gather + A-fragment packs ----------
__global__ void k_pack_rnn(const float* __restrict__ emb, const int* __restrict__ tgt,
                           const float* __restrict__ ctx, unsigned short* __restrict__ frag,
                           int t) {
  int tid = blockIdx.x * blockDim.x + threadIdx.x;
  if (tid >= BB * KIH) return;
  int m, k, mt;
  packA_decomp(tid, KIH / 32, m, k, mt);
  float val;
  if (k < EE) { int tok = tgt[m * TT + t]; val = emb[(size_t)tok * EE + k]; }
  else        { val = ctx[m * HH + (k - EE)]; }
  frag[tid] = f2bf(val);
}

__global__ void k_pack_h(const float* __restrict__ h, unsigned short* __restrict__ frag) {
  int tid = blockIdx.x * blockDim.x + threadIdx.x;
  if (tid >= BB * HH) return;
  int m, k, mt;
  packA_decomp(tid, HH / 32, m, k, mt);
  frag[tid] = f2bf(h[m * HH + k]);
}

__global__ void k_pack_pred(const float* __restrict__ ctx, const float* __restrict__ h,
                            unsigned short* __restrict__ frag) {
  int tid = blockIdx.x * blockDim.x + threadIdx.x;
  if (tid >= BB * K2H) return;
  int m, k, mt;
  packA_decomp(tid, K2H / 32, m, k, mt);
  frag[tid] = f2bf((k < HH) ? ctx[m * HH + k] : h[m * HH + (k - HH)]);
}

// ---------- h0 = init_h @ W_map^T + b_map ----------
__global__ void k_hidden_map(const float* __restrict__ ih, const float* __restrict__ Wm,
                             const float* __restrict__ bm, float* __restrict__ h) {
  int tid = blockIdx.x * blockDim.x + threadIdx.x;
  if (tid >= BB * HH) return;
  int m = tid / HH, n = tid % HH;
  float acc = bm[n];
  const float* xr = ih + (size_t)m * HH;
  const float* wr = Wm + (size_t)n * HH;
  for (int k = 0; k < HH; ++k) acc += xr[k] * wr[k];
  h[tid] = acc;
}

__global__ void k_zero(float* __restrict__ p, int n) {
  int tid = blockIdx.x * blockDim.x + threadIdx.x;
  if (tid < n) p[tid] = 0.0f;
}

// ---------- GRU gate GEMMs: gi = rnn_in@Wih^T + b_ih, gh = h@Whh^T + b_hh ----------
// grid.x = 192 N-tiles, grid.y = {0:gi, 1:gh}; 64 threads = 2 waves, wave = M-tile.
// Kept wave-per-tile: GRU is on the serial critical path -> maximize parallel waves,
// minimize per-wave WMMA chain length.
__global__ void k_gru_gemm(const unsigned short* __restrict__ fragRnn,
                           const unsigned short* __restrict__ fragH,
                           const unsigned short* __restrict__ Wih,
                           const unsigned short* __restrict__ Whh,
                           const float* __restrict__ b_ih, const float* __restrict__ b_hh,
                           float* __restrict__ gi, float* __restrict__ gh) {
  const int nt   = blockIdx.x;
  const int lane = threadIdx.x & 31;
  const int mt   = threadIdx.x >> 5;
  const unsigned short* fa; const unsigned short* fb; const float* bias; float* out; int KB;
  if (blockIdx.y == 0) { fa = fragRnn; fb = Wih; bias = b_ih; out = gi; KB = KIH / 32; }
  else                 { fa = fragH;   fb = Whh; bias = b_hh; out = gh; KB = HH  / 32; }
  const unsigned short* pa = fa + ((size_t)(mt * KB) * 32 + lane) * 16;
  const unsigned short* pb = fb + ((size_t)(nt * KB) * 32 + lane) * 16;
  v8f c = {};
  for (int kb = 0; kb < KB; ++kb) {
    __builtin_prefetch(pb + 512, 0, 0);
    v16bf a  = *reinterpret_cast<const v16bf*>(pa);
    v16bf bm = *reinterpret_cast<const v16bf*>(pb);
    c = WMMA_BF16(a, bm, c);
    pa += 512; pb += 512;
  }
  const int n  = nt * 16 + (lane & 15);
  const float bn = bias[n];
  const int mbase = mt * 16 + ((lane >> 4) << 3);   // C/D: VGPR v -> M = v + 8*(lane>=16)
#pragma unroll
  for (int v = 0; v < 8; ++v)
    out[(size_t)(mbase + v) * G3H + n] = c[v] + bn;
}

// ---------- GRU gate nonlinearity ----------
__global__ void k_gru_update(const float* __restrict__ gi, const float* __restrict__ gh,
                             float* __restrict__ h) {
  int tid = blockIdx.x * blockDim.x + threadIdx.x;
  if (tid >= BB * HH) return;
  int m = tid / HH, j = tid % HH;
  size_t base = (size_t)m * G3H;
  float r  = sigm(gi[base + j]          + gh[base + j]);
  float z  = sigm(gi[base + HH + j]     + gh[base + HH + j]);
  float nn = tanhf(gi[base + 2*HH + j]  + r * gh[base + 2*HH + j]);
  h[tid] = (1.0f - z) * nn + z * h[tid];
}

// ---------- dot-product attention + softmax + context ----------
__global__ void k_attn(const float* __restrict__ enc, const float* __restrict__ h,
                       float* __restrict__ ctx) {
  const int b = blockIdx.x;
  const int tid = threadIdx.x;                 // 256 threads
  __shared__ float sh[HH];
  __shared__ float ps[256];
  __shared__ float sc[SS];
  __shared__ float sinv;
  for (int i = tid; i < HH; i += 256) sh[i] = h[b * HH + i];
  __syncthreads();
  {
    int s = tid & 63, part = tid >> 6;         // 4 partials x 64 scores
    const float* er = enc + ((size_t)b * SS + s) * HH + part * 256;
    const float* hr = sh + part * 256;
    float acc = 0.0f;
    for (int k = 0; k < 256; ++k) acc += er[k] * hr[k];
    ps[tid] = acc;
  }
  __syncthreads();
  if (tid < SS) sc[tid] = ps[tid] + ps[tid + 64] + ps[tid + 128] + ps[tid + 192];
  __syncthreads();
  if (tid == 0) {
    float mx = sc[0];
    for (int s = 1; s < SS; ++s) mx = fmaxf(mx, sc[s]);
    float sum = 0.0f;
    for (int s = 0; s < SS; ++s) { float e = expf(sc[s] - mx); sc[s] = e; sum += e; }
    sinv = 1.0f / sum;
  }
  __syncthreads();
  for (int j = tid; j < HH; j += 256) {
    float acc = 0.0f;
    for (int s = 0; s < SS; ++s) acc += sc[s] * enc[((size_t)b * SS + s) * HH + j];
    ctx[b * HH + j] = acc * sinv;
  }
}

// ---------- classifier: logits = pred @ Wcls^T + b_cls ----------
// 250 blocks x 128 threads (4 waves); each wave handles 2 N-tiles x both M-tiles.
// Each A-fragment pair feeds 4 WMMAs (2 b128 loads per WMMA); A L2 traffic
// halves to ~131 MB/step, matching the bf16 Wcls stream which is L2-resident
// (131 MB < 192 MB L2 on MI455X).
__global__ void k_cls(const unsigned short* __restrict__ fragPred,
                      const unsigned short* __restrict__ Wcls,
                      const float* __restrict__ b_cls, float* __restrict__ out, int t) {
  const int lane = threadIdx.x & 31;
  const int wv   = threadIdx.x >> 5;                 // 0..3
  const int nt0  = (blockIdx.x * 4 + wv) * 2;        // first of 2 N-tiles
  const int KB = K2H / 32;                           // 64
  const unsigned short* pa0 = fragPred + (size_t)lane * 16;                // M-tile 0
  const unsigned short* pa1 = fragPred + ((size_t)(KB * 32) + lane) * 16;  // M-tile 1
  const unsigned short* pb0 = Wcls + ((size_t)nt0 * KB * 32 + lane) * 16;
  const unsigned short* pb1 = pb0 + (size_t)KB * 512;
  v8f c00 = {}, c10 = {}, c01 = {}, c11 = {};
  for (int kb = 0; kb < KB; ++kb) {
    __builtin_prefetch(pb0 + 512, 0, 0);
    __builtin_prefetch(pb1 + 512, 0, 0);
    v16bf a0 = *reinterpret_cast<const v16bf*>(pa0);
    v16bf a1 = *reinterpret_cast<const v16bf*>(pa1);
    v16bf b0 = *reinterpret_cast<const v16bf*>(pb0);
    v16bf b1 = *reinterpret_cast<const v16bf*>(pb1);
    c00 = WMMA_BF16(a0, b0, c00);
    c10 = WMMA_BF16(a1, b0, c10);
    c01 = WMMA_BF16(a0, b1, c01);
    c11 = WMMA_BF16(a1, b1, c11);
    pa0 += 512; pa1 += 512; pb0 += 512; pb1 += 512;
  }
  const int g8 = (lane >> 4) << 3;
#pragma unroll
  for (int j = 0; j < 2; ++j) {
    const int n = (nt0 + j) * 16 + (lane & 15);
    const float bn = b_cls[n];
    v8f c0 = j ? c01 : c00;
    v8f c1 = j ? c11 : c10;
#pragma unroll
    for (int v = 0; v < 8; ++v) {
      int r0 = g8 + v;                               // batch rows 0..15
      int r1 = 16 + g8 + v;                          // batch rows 16..31
      out[((size_t)r0 * TT + t) * VV + n] = c0[v] + bn;
      out[((size_t)r1 * TT + t) * VV + n] = c1[v] + bn;
    }
  }
}

// ---------- host ----------
extern "C" void kernel_launch(void* const* d_in, const int* in_sizes, int n_in,
                              void* d_out, int out_size, void* d_ws, size_t ws_size,
                              hipStream_t stream) {
  const float* enc    = (const float*)d_in[0];
  const float* init_h = (const float*)d_in[1];
  const int*   tgt    = (const int*)  d_in[2];
  const float* emb    = (const float*)d_in[3];
  const float* W_ih   = (const float*)d_in[4];
  const float* b_ih   = (const float*)d_in[5];
  const float* W_hh   = (const float*)d_in[6];
  const float* b_hh   = (const float*)d_in[7];
  const float* W_map  = (const float*)d_in[8];
  const float* b_map  = (const float*)d_in[9];
  const float* W_cls  = (const float*)d_in[10];
  const float* b_cls  = (const float*)d_in[11];
  float* out = (float*)d_out;

  char* ws = (char*)d_ws;
  size_t off = 0;
  auto take = [&](size_t bytes) -> char* {
    char* p = ws + off;
    off = (off + bytes + 255) & ~(size_t)255;
    return p;
  };
  unsigned short* WclsSw  = (unsigned short*)take((size_t)VV  * K2H * 2);
  unsigned short* WihSw   = (unsigned short*)take((size_t)G3H * KIH * 2);
  unsigned short* WhhSw   = (unsigned short*)take((size_t)G3H * HH  * 2);
  float* h   = (float*)take((size_t)BB * HH  * 4);
  float* ctx = (float*)take((size_t)BB * HH  * 4);
  float* gi  = (float*)take((size_t)BB * G3H * 4);
  float* gh  = (float*)take((size_t)BB * G3H * 4);
  unsigned short* fragRnn  = (unsigned short*)take((size_t)BB * KIH * 2);
  unsigned short* fragH    = (unsigned short*)take((size_t)BB * HH  * 2);
  unsigned short* fragPred = (unsigned short*)take((size_t)BB * K2H * 2);
  (void)ws_size; (void)n_in; (void)in_sizes; (void)out_size;

  auto blks = [](int total) { return (total + 255) / 256; };

  // one-time weight bf16 swizzle (deterministic; redone every launch)
  { int tot = VV  * K2H; k_swizzleB<<<blks(tot), 256, 0, stream>>>(W_cls, WclsSw, K2H/32, tot); }
  { int tot = G3H * KIH; k_swizzleB<<<blks(tot), 256, 0, stream>>>(W_ih,  WihSw,  KIH/32, tot); }
  { int tot = G3H * HH;  k_swizzleB<<<blks(tot), 256, 0, stream>>>(W_hh,  WhhSw,  HH/32,  tot); }
  k_hidden_map<<<blks(BB*HH), 256, 0, stream>>>(init_h, W_map, b_map, h);
  k_zero<<<blks(BB*HH), 256, 0, stream>>>(ctx, BB*HH);

  for (int t = 0; t < TT; ++t) {
    k_pack_rnn<<<blks(BB*KIH), 256, 0, stream>>>(emb, tgt, ctx, fragRnn, t);
    k_pack_h<<<blks(BB*HH), 256, 0, stream>>>(h, fragH);
    dim3 gg(G3H / 16, 2);
    k_gru_gemm<<<gg, 64, 0, stream>>>(fragRnn, fragH, WihSw, WhhSw, b_ih, b_hh, gi, gh);
    k_gru_update<<<blks(BB*HH), 256, 0, stream>>>(gi, gh, h);
    k_attn<<<BB, 256, 0, stream>>>(enc, h, ctx);
    k_pack_pred<<<blks(BB*K2H), 256, 0, stream>>>(ctx, h, fragPred);
    k_cls<<<VV / 128, 128, 0, stream>>>(fragPred, WclsSw, b_cls, out, t);
  }
}
